// AtomicSidechainEncoder_32933809226362
// MI455X (gfx1250) — matrix operand order repaired and checked by reading, not compiled, not run on hardware
//
#include <hip/hip_runtime.h>

// ---------------------------------------------------------------------------
// AtomicSidechainEncoder for MI455X (gfx1250, wave32, WMMA + TDM).
// Heavy GEMMs (64x64 per_l_linear, edge linears, attention logits) run on
// v_wmma_f32_16x16x32_bf16 with fp32 accumulation; weights are pre-packed
// into per-lane bf16 B fragments matching the ISA 7.12.2 wave32 layout.
// The message and edge-update kernels stage their weight fragments into LDS
// with the Tensor Data Mover (tensor_load_to_lds + s_wait_tensorcnt).
// Segment softmax uses flipped-uint atomicMax + atomicAdd; message scatter
// uses global_atomic_add_f32. Sigmoid/softmax use v_rcp_f32 (no IEEE div).
// ---------------------------------------------------------------------------

#define NRES    4096
#define APR     8
#define NATOMS  (NRES*APR)        // 32768
#define ERAD    262144
#define KNN     30
#define ERES    (NRES*KNN)        // 122880
#define CCH     64
#define NH      8
#define NLAY    4

typedef __attribute__((ext_vector_type(16))) __bf16 v16bf;
typedef __attribute__((ext_vector_type(8)))  float  v8f;
typedef __attribute__((ext_vector_type(4)))  unsigned v4u;
typedef __attribute__((ext_vector_type(8)))  int    v8i;
typedef __attribute__((ext_vector_type(4)))  int    v4i;

union AB { v16bf v; unsigned u[8]; unsigned short s[16]; };

#if defined(__has_builtin)
#if __has_builtin(__builtin_amdgcn_tensor_load_to_lds)
#define HAVE_TDM 1
#endif
#endif
#ifndef HAVE_TDM
#define HAVE_TDM 0
#endif
#if __has_include(<hip/amd_detail/amd_gfx1250_TDM.h>)
#define TDM_ARITY6 1   // therock-10.0 headers -> 6-arg builtin
#else
#define TDM_ARITY6 0   // ROCm 7.2 -> 5-arg builtin
#endif

static __device__ inline unsigned short f2bf(float x) {
    unsigned b = __float_as_uint(x);
    b += 0x7FFFu + ((b >> 16) & 1u);          // round-to-nearest-even
    return (unsigned short)(b >> 16);
}

static __device__ inline v8f zero8() {
    v8f z;
#pragma unroll
    for (int i = 0; i < 8; ++i) z[i] = 0.f;
    return z;
}

static __device__ inline v8f wmma_bf(v16bf a, v16bf b, v8f c) {
    // D = A(16x32 bf16) * B(32x16 bf16) + C(16x16 f32)
    return __builtin_amdgcn_wmma_f32_16x16x32_bf16(false, a, false, b,
                                                   (short)0, c, false, false);
}

// A fragment from a row-major bf16 panel [16][kstride] in LDS.
// Lanes 0-15: row=lane, K = k0+{0..7, 16..23}; lanes 16-31: K = k0+{8..15, 24..31}.
static __device__ inline v16bf load_a_frag(const unsigned short* panel,
                                           int kstride, int k0, int lane) {
    AB r;
    const unsigned short* p = panel + (lane & 15) * kstride + k0 + ((lane >> 4) & 1) * 8;
#pragma unroll
    for (int i = 0; i < 8; ++i) {
        int eo = 2 * (i & 3) + ((i & 4) ? 16 : 0);
        r.s[2 * i]     = p[eo];
        r.s[2 * i + 1] = p[eo + 1];
    }
    return r.v;
}

// B fragment: pre-packed [kstep][ntile][lane][8 dwords] in global memory.
static __device__ inline v16bf load_b_frag(const unsigned* __restrict__ packed,
                                           int ntiles, int ks, int t, int lane) {
    AB r;
    const uint4* q = (const uint4*)(packed + (((size_t)(ks * ntiles + t)) * 32 + lane) * 8);
    uint4 a = q[0], b = q[1];
    r.u[0] = a.x; r.u[1] = a.y; r.u[2] = a.z; r.u[3] = a.w;
    r.u[4] = b.x; r.u[5] = b.y; r.u[6] = b.z; r.u[7] = b.w;
    return r.v;
}

// Same layout, but fragments resident in LDS (ds_load path after TDM copy).
static __device__ inline v16bf load_b_frag_lds(const unsigned* base,
                                               int ntiles, int ks, int t, int lane) {
    AB r;
    const unsigned* p = base + (((ks * ntiles) + t) * 32 + lane) * 8;
#pragma unroll
    for (int i = 0; i < 8; ++i) r.u[i] = p[i];
    return r.v;
}

struct Frags { v16bf a[4][2]; };   // 4 M-panels (l0 + 3 l1 rows), 2 K-steps

static __device__ inline void load_frags(const unsigned short P[][16][CCH],
                                         int lane, Frags& f) {
#pragma unroll
    for (int l = 0; l < 4; ++l) {
        f.a[l][0] = load_a_frag(&P[l][0][0], CCH, 0,  lane);
        f.a[l][1] = load_a_frag(&P[l][0][0], CCH, 32, lane);
    }
}

// One N-tile of the per_l_linear: out[0] = A0*W0, out[1..3] = A1_i*W1.
static __device__ inline void gemm_t(const Frags& f, const unsigned* W0p,
                                     const unsigned* W1p, int t, int lane, v8f out[4]) {
    v16bf b00 = load_b_frag(W0p, 4, 0, t, lane);
    v16bf b01 = load_b_frag(W0p, 4, 1, t, lane);
    v16bf b10 = load_b_frag(W1p, 4, 0, t, lane);
    v16bf b11 = load_b_frag(W1p, 4, 1, t, lane);
    out[0] = wmma_bf(f.a[0][0], b00, zero8());
    out[0] = wmma_bf(f.a[0][1], b01, out[0]);
#pragma unroll
    for (int l = 1; l < 4; ++l) {
        out[l] = wmma_bf(f.a[l][0], b10, zero8());
        out[l] = wmma_bf(f.a[l][1], b11, out[l]);
    }
}

static __device__ inline void gemm_t_lds(const Frags& f, const unsigned* W0l,
                                         const unsigned* W1l, int t, int lane, v8f out[4]) {
    v16bf b00 = load_b_frag_lds(W0l, 4, 0, t, lane);
    v16bf b01 = load_b_frag_lds(W0l, 4, 1, t, lane);
    v16bf b10 = load_b_frag_lds(W1l, 4, 0, t, lane);
    v16bf b11 = load_b_frag_lds(W1l, 4, 1, t, lane);
    out[0] = wmma_bf(f.a[0][0], b00, zero8());
    out[0] = wmma_bf(f.a[0][1], b01, out[0]);
#pragma unroll
    for (int l = 1; l < 4; ++l) {
        out[l] = wmma_bf(f.a[l][0], b10, zero8());
        out[l] = wmma_bf(f.a[l][1], b11, out[l]);
    }
}

static __device__ inline unsigned flipf(float f) {
    unsigned b = __float_as_uint(f);
    return (b & 0x80000000u) ? ~b : (b | 0x80000000u);
}
static __device__ inline float unflipf(unsigned u) {
    return __uint_as_float((u & 0x80000000u) ? (u ^ 0x80000000u) : ~u);
}
// Fast sigmoid: single v_rcp_f32 instead of the IEEE division chain.
static __device__ inline float sigm(float x) {
    return __builtin_amdgcn_rcpf(1.f + __expf(-x));
}

// ---------------------------------------------------------------------------
// TDM: copy `dwords` dwords from global `src` to LDS byte offset `lds_off`.
// D# per ISA 8.3/8.4: group0 {count=1, lds_addr, global_addr, type=2},
// group1 {data_size=4B, 1-row tile of `dwords` elements}.
// ---------------------------------------------------------------------------
static __device__ inline void set_bits(unsigned* w, int lo, int len,
                                       unsigned long long val) {
    for (int i = 0; i < len; ++i)
        if ((val >> i) & 1ull) w[(lo + i) >> 5] |= 1u << ((lo + i) & 31);
}

static __device__ inline void tdm_copy_to_lds(const void* src, unsigned lds_off,
                                              unsigned dwords) {
#if HAVE_TDM
    unsigned g0w[4] = {0, 0, 0, 0};
    unsigned g1w[8] = {0, 0, 0, 0, 0, 0, 0, 0};
    unsigned long long ga = (unsigned long long)(size_t)src;
    set_bits(g0w, 0, 2, 1ull);              // count = 1 (valid descriptor)
    set_bits(g0w, 32, 32, lds_off);         // lds_addr (bytes)
    set_bits(g0w, 64, 57, ga);              // global_addr (bytes)
    set_bits(g0w, 126, 2, 2ull);            // type = 2 ("image")
    set_bits(g1w, 16, 2, 2ull);             // data_size = 4 bytes
    set_bits(g1w, 48, 32, dwords);          // tensor_dim0
    set_bits(g1w, 80, 32, 1ull);            // tensor_dim1
    set_bits(g1w, 112, 16, dwords);         // tile_dim0
    set_bits(g1w, 128, 16, 1ull);           // tile_dim1
    set_bits(g1w, 160, 48, dwords);         // tensor_dim0_stride
    v4u g0;
    g0.x = (unsigned)__builtin_amdgcn_readfirstlane((int)g0w[0]);
    g0.y = (unsigned)__builtin_amdgcn_readfirstlane((int)g0w[1]);
    g0.z = (unsigned)__builtin_amdgcn_readfirstlane((int)g0w[2]);
    g0.w = (unsigned)__builtin_amdgcn_readfirstlane((int)g0w[3]);
    v8i g1;
#pragma unroll
    for (int i = 0; i < 8; ++i)
        g1[i] = __builtin_amdgcn_readfirstlane((int)g1w[i]);
    v4i gz;
#pragma unroll
    for (int i = 0; i < 4; ++i) gz[i] = 0;
#if TDM_ARITY6
    v8i gz8;
#pragma unroll
    for (int i = 0; i < 8; ++i) gz8[i] = 0;
    __builtin_amdgcn_tensor_load_to_lds(g0, g1, gz, gz, gz8, 0);
#else
    __builtin_amdgcn_tensor_load_to_lds(g0, g1, gz, gz, 0);
#endif
    __builtin_amdgcn_s_wait_tensorcnt(0);
#else
    (void)src; (void)lds_off; (void)dwords;
#endif
}

// ---------------------------------------------------------------------------
// Weight packing: W[K][Nr] f32 row-major -> bf16 B fragments, N padded to Np.
// ---------------------------------------------------------------------------
__global__ void k_pack_w(const float* __restrict__ W, int K, int Nr, int Np,
                         unsigned* __restrict__ dst) {
    int ksteps = (K + 31) >> 5, ntiles = Np >> 4;
    int total = ksteps * ntiles * 256;
    for (int idx = blockIdx.x * blockDim.x + threadIdx.x; idx < total;
         idx += gridDim.x * blockDim.x) {
        int v    = idx & 7;
        int lane = (idx >> 3) & 31;
        int rest = idx >> 8;
        int t    = rest % ntiles;
        int ks   = rest / ntiles;
        int n    = t * 16 + (lane & 15);
        int k0   = ks * 32 + ((lane & 16) ? 8 : 0) + ((v & 4) ? 16 : 0) + 2 * (v & 3);
        float w0 = (k0     < K && n < Nr) ? W[(size_t)k0 * Nr + n]       : 0.f;
        float w1 = (k0 + 1 < K && n < Nr) ? W[(size_t)(k0 + 1) * Nr + n] : 0.f;
        dst[idx] = (unsigned)f2bf(w0) | ((unsigned)f2bf(w1) << 16);
    }
}

__global__ void k_fill(unsigned* p, unsigned v, int n) {
    int i = blockIdx.x * blockDim.x + threadIdx.x;
    if (i < n) p[i] = v;
}

// ---------------------------------------------------------------------------
// Atom embedding: x[:,0,:] = emb @ W0 + b, l=1..3 zeroed.
// ---------------------------------------------------------------------------
__global__ void k_embed(const float* __restrict__ emb, const float* __restrict__ W,
                        const float* __restrict__ b, float* __restrict__ x) {
    int a = blockIdx.x, c = threadIdx.x;
    float acc = b[c];
#pragma unroll
    for (int k = 0; k < 25; ++k) acc += emb[(size_t)a * 25 + k] * W[k * 64 + c];
    float* o = x + (size_t)a * 256;
    o[c] = acc; o[64 + c] = 0.f; o[128 + c] = 0.f; o[192 + c] = 0.f;
}

// ---------------------------------------------------------------------------
// Equivariant RMS norm: xn = concat(x0/r0*g0, x1/r1*g1).
// ---------------------------------------------------------------------------
__global__ void k_eqnorm(const float* __restrict__ x, const float* __restrict__ g,
                         float* __restrict__ xn) {
    int n = blockIdx.x, c = threadIdx.x;
    __shared__ float red[64];
    const float* xr = x + (size_t)n * 256;
    float x0 = xr[c], x1 = xr[64 + c], x2 = xr[128 + c], x3 = xr[192 + c];
    red[c] = x0 * x0; __syncthreads();
    for (int s = 32; s > 0; s >>= 1) { if (c < s) red[c] += red[c + s]; __syncthreads(); }
    float s0 = red[0]; __syncthreads();
    red[c] = x1 * x1 + x2 * x2 + x3 * x3; __syncthreads();
    for (int s = 32; s > 0; s >>= 1) { if (c < s) red[c] += red[c + s]; __syncthreads(); }
    float s1 = red[0];
    float i0 = rsqrtf(s0 * (1.f / 64.f) + 1e-6f);
    float i1 = rsqrtf(s1 * (1.f / 192.f) + 1e-6f);
    float* o = xn + (size_t)n * 256;
    o[c]       = x0 * i0 * g[c];
    o[64 + c]  = x1 * i1 * g[64 + c];
    o[128 + c] = x2 * i1 * g[64 + c];
    o[192 + c] = x3 * i1 * g[64 + c];
}

// ---------------------------------------------------------------------------
// Edge geometry: Wigner-D (l=1 block) + RBF (mode 0: rbf32, mode 1: rbf16+pos16).
// ---------------------------------------------------------------------------
__global__ void k_edge_geom(const float* __restrict__ pos, const int* __restrict__ ei,
                            int E, int mode, float* __restrict__ D1,
                            float* __restrict__ e) {
    int i = blockIdx.x * blockDim.x + threadIdx.x;
    if (i >= E) return;
    int s = ei[i], d = ei[E + i];
    float vx = pos[d * 3] - pos[s * 3];
    float vy = pos[d * 3 + 1] - pos[s * 3 + 1];
    float vz = pos[d * 3 + 2] - pos[s * 3 + 2];
    float dist = sqrtf(vx * vx + vy * vy + vz * vz);
    float inv = __builtin_amdgcn_rcpf(dist + 1e-8f);
    float nx = vx * inv, ny = vy * inv, nz = vz * inv;
    float rx, ry, rz;
    if (fabsf(nz) < 0.99f) { rx = 0.f; ry = 0.f; rz = 1.f; }
    else                   { rx = 1.f; ry = 0.f; rz = 0.f; }
    float ux = ny * rz - nz * ry, uy = nz * rx - nx * rz, uz = nx * ry - ny * rx;
    float un = __builtin_amdgcn_rcpf(sqrtf(ux * ux + uy * uy + uz * uz) + 1e-8f);
    ux *= un; uy *= un; uz *= un;
    float wx = ny * uz - nz * uy, wy = nz * ux - nx * uz, wz = nx * uy - ny * ux;
    float R[3][3] = {{ux, uy, uz}, {nx, ny, nz}, {wx, wy, wz}};
    const int perm[3] = {1, 2, 0};
#pragma unroll
    for (int a = 0; a < 3; ++a)
#pragma unroll
        for (int b = 0; b < 3; ++b)
            D1[(size_t)i * 9 + a * 3 + b] = R[perm[a]][perm[b]];
    float* eo = e + (size_t)i * 32;
    if (mode == 0) {
        const float isg = 32.f / 20.f;
#pragma unroll
        for (int k = 0; k < 32; ++k) {
            float mu = 20.f * (float)k / 31.f;
            float z = (dist - mu) * isg;
            eo[k] = __expf(-z * z);
        }
    } else {
        const float isg = 16.f / 20.f;
#pragma unroll
        for (int k = 0; k < 16; ++k) {
            float mu = 20.f * (float)k / 15.f;
            float z = (dist - mu) * isg;
            eo[k] = __expf(-z * z);
        }
        float dd = (float)(s - d);
#pragma unroll
        for (int q = 0; q < 8; ++q) {
            float freq = __expf(-logf(10000.f) / 16.f * (float)(2 * q));
            float ang = dd * freq;
            eo[16 + q] = cosf(ang);
            eo[24 + q] = sinf(ang);
        }
    }
}

// ---------------------------------------------------------------------------
// Pass A: s = silu(e @ edge_w + b) via WMMA, then attention logits via WMMA
// on the invariant panel [16, 192] @ alpha_w[192, 8->16]; atomicMax seg-max.
// 1 wave per 16-edge tile, 4 waves / block.
// ---------------------------------------------------------------------------
__global__ void __launch_bounds__(128)
k_edge_s_logits(const float* __restrict__ efeat, const float* __restrict__ xn,
                const int* __restrict__ ei, int E,
                const unsigned* __restrict__ ewp, const float* __restrict__ eb,
                const unsigned* __restrict__ awp,
                float* __restrict__ sbuf, float* __restrict__ logits,
                unsigned* __restrict__ lmax) {
    int wave = threadIdx.x >> 5, lane = threadIdx.x & 31;
    int e0 = (blockIdx.x * 4 + wave) * 16;
    __shared__ unsigned short ebuf[4][16][32];
    __shared__ unsigned short inv[4][16][192];   // [xn_src0 | xn_dst0 | s]
    for (int i = lane; i < 16 * 32; i += 32) {
        int e = i >> 5, k = i & 31;
        ebuf[wave][e][k] = f2bf(efeat[(size_t)(e0 + e) * 32 + k]);
    }
    for (int i = lane; i < 16 * 64; i += 32) {
        int e = i >> 6, c = i & 63;
        int s = ei[e0 + e], d = ei[E + e0 + e];
        inv[wave][e][c]      = f2bf(xn[(size_t)s * 256 + c]);
        inv[wave][e][64 + c] = f2bf(xn[(size_t)d * 256 + c]);
    }
    __syncthreads();
    v16bf a = load_a_frag(&ebuf[wave][0][0], 32, 0, lane);
    int col = lane & 15, hi = (lane >> 4) & 1;
#pragma unroll
    for (int t = 0; t < 4; ++t) {
        v8f acc = wmma_bf(a, load_b_frag(ewp, 4, 0, t, lane), zero8());
#pragma unroll
        for (int r = 0; r < 8; ++r) {
            int e = r + hi * 8, c = t * 16 + col;
            float v = acc[r] + eb[c];
            v = v * sigm(v);
            sbuf[(size_t)(e0 + e) * 64 + c] = v;
            inv[wave][e][128 + c] = f2bf(v);
        }
    }
    __syncthreads();
    v8f lac = zero8();
#pragma unroll
    for (int ks = 0; ks < 6; ++ks) {
        v16bf af = load_a_frag(&inv[wave][0][0], 192, ks * 32, lane);
        lac = wmma_bf(af, load_b_frag(awp, 1, ks, 0, lane), lac);
    }
    if (col < NH) {
#pragma unroll
        for (int r = 0; r < 8; ++r) {
            int e = r + hi * 8;
            float l = lac[r];
            l = l > 0.f ? l : 0.2f * l;
            logits[(size_t)(e0 + e) * 8 + col] = l;
            atomicMax(&lmax[(size_t)ei[E + e0 + e] * 8 + col], flipf(l));
        }
    }
}

// Pass B: el = exp(logit - max), atomicAdd denominator. Done in place.
__global__ void k_edge_expsum(float* __restrict__ lg, const unsigned* __restrict__ lmax,
                              const int* __restrict__ ei, int E, float* __restrict__ den) {
    int idx = blockIdx.x * blockDim.x + threadIdx.x;
    if (idx >= E * 8) return;
    int e = idx >> 3, h = idx & 7;
    int d = ei[E + e];
    float el = __expf(lg[idx] - unflipf(lmax[(size_t)d * 8 + h]));
    lg[idx] = el;
    atomicAdd(&den[(size_t)d * 8 + h], el);
}

// ---------------------------------------------------------------------------
// Pass C: rotate source into edge frame, per_l_linear(w_msg) on WMMA with
// TDM-staged weight fragments in LDS, scale by s * alpha, rotate back (D^T),
// atomic scatter into agg[dst].
// ---------------------------------------------------------------------------
__global__ void __launch_bounds__(128)
k_edge_message(const float* __restrict__ xn, const int* __restrict__ ei, int E,
               const float* __restrict__ D1, const float* __restrict__ sbuf,
               const float* __restrict__ el, const float* __restrict__ den,
               const unsigned* __restrict__ wm0, const unsigned* __restrict__ wm1,
               float* __restrict__ agg) {
    int wave = threadIdx.x >> 5, lane = threadIdx.x & 31;
    int e0 = (blockIdx.x * 4 + wave) * 16;
    __shared__ unsigned short P[4][4][16][CCH];   // rotated A panels (l0, 3x l1)
    __shared__ float Dt[4][16][9];
    __shared__ float al[4][16][8];
    __shared__ int   dstn[4][16];
    __shared__ unsigned wbuf[4096];               // 16KB: wm0|wm1 B fragments

    // Stage both weight-fragment buffers (contiguous in ws) into LDS via TDM.
#if HAVE_TDM
    if (threadIdx.x < 32)
        tdm_copy_to_lds(wm0, (unsigned)(size_t)(void*)wbuf, 4096);
#else
    for (int i = threadIdx.x; i < 4096; i += 128) wbuf[i] = wm0[i];
#endif
    // prefetch next tile's scalar stream
    __builtin_prefetch((const void*)(sbuf + (size_t)(e0 + 64) * 64), 0, 0);
    for (int i = lane; i < 16; i += 32) dstn[wave][i] = ei[E + e0 + i];
    for (int i = lane; i < 16 * 9; i += 32)
        Dt[wave][i / 9][i % 9] = D1[(size_t)(e0 + i / 9) * 9 + (i % 9)];
    __syncthreads();
    for (int i = lane; i < 16 * 8; i += 32) {
        int e = i >> 3, h = i & 7;
        al[wave][e][h] = el[(size_t)(e0 + e) * 8 + h] *
                         __builtin_amdgcn_rcpf(den[(size_t)dstn[wave][e] * 8 + h] + 1e-9f);
    }
    for (int i = lane; i < 16 * 64; i += 32) {
        int e = i >> 6, c = i & 63;
        int s = ei[e0 + e];
        const float* xr = xn + (size_t)s * 256 + c;
        float x0 = xr[0], x1 = xr[64], x2 = xr[128], x3 = xr[192];
        const float* De = &Dt[wave][e][0];
        P[wave][0][e][c] = f2bf(x0);
        P[wave][1][e][c] = f2bf(De[0] * x1 + De[1] * x2 + De[2] * x3);
        P[wave][2][e][c] = f2bf(De[3] * x1 + De[4] * x2 + De[5] * x3);
        P[wave][3][e][c] = f2bf(De[6] * x1 + De[7] * x2 + De[8] * x3);
    }
    __syncthreads();
    Frags fr;
    load_frags(P[wave], lane, fr);
    int col = lane & 15, hi = (lane >> 4) & 1;
#pragma unroll
    for (int t = 0; t < 4; ++t) {
        v8f m[4];
        gemm_t_lds(fr, wbuf, wbuf + 2048, t, lane, m);
#pragma unroll
        for (int r = 0; r < 8; ++r) {
            int e = r + hi * 8, c = t * 16 + col;
            float sa = sbuf[(size_t)(e0 + e) * 64 + c] * al[wave][e][c >> 3];
            float v0 = m[0][r] * sa, v1 = m[1][r] * sa;
            float v2 = m[2][r] * sa, v3 = m[3][r] * sa;
            const float* De = &Dt[wave][e][0];
            float o1 = De[0] * v1 + De[3] * v2 + De[6] * v3;
            float o2 = De[1] * v1 + De[4] * v2 + De[7] * v3;
            float o3 = De[2] * v1 + De[5] * v2 + De[8] * v3;
            float* ap = agg + (size_t)dstn[wave][e] * 256 + c;
            atomicAdd(ap, v0);
            atomicAdd(ap + 64, o1);
            atomicAdd(ap + 128, o2);
            atomicAdd(ap + 192, o3);
        }
    }
}

// ---------------------------------------------------------------------------
// Node update: x += per_l(agg,w_out); eq_norm(ln2); gated FFN (2 GEMMs).
// 1 wave per 16-node tile. 3 WMMA GEMMs back-to-back.
// ---------------------------------------------------------------------------
__global__ void __launch_bounds__(128)
k_node_update(const float* __restrict__ x_in, float* __restrict__ x_out,
              const float* __restrict__ agg,
              const unsigned* __restrict__ wo0, const unsigned* __restrict__ wo1,
              const float* __restrict__ ln2,
              const unsigned* __restrict__ f10, const unsigned* __restrict__ f11,
              const unsigned* __restrict__ f20, const unsigned* __restrict__ f21) {
    int wave = threadIdx.x >> 5, lane = threadIdx.x & 31;
    int n0 = (blockIdx.x * 4 + wave) * 16;
    __shared__ float xb[4][16][4][CCH];           // running x for the tile
    __shared__ unsigned short P[4][4][16][CCH];   // bf16 A panels
    __shared__ float nrm[4][2][16];
    int col = lane & 15, hi = (lane >> 4) & 1;

    for (int i = lane; i < 16 * 64; i += 32) {
        int n = i >> 6, c = i & 63;
        const float* xr = x_in + (size_t)(n0 + n) * 256 + c;
        const float* ar = agg  + (size_t)(n0 + n) * 256 + c;
        xb[wave][n][0][c] = xr[0];   xb[wave][n][1][c] = xr[64];
        xb[wave][n][2][c] = xr[128]; xb[wave][n][3][c] = xr[192];
        P[wave][0][n][c] = f2bf(ar[0]);   P[wave][1][n][c] = f2bf(ar[64]);
        P[wave][2][n][c] = f2bf(ar[128]); P[wave][3][n][c] = f2bf(ar[192]);
    }
    __syncthreads();

    Frags fr;
    load_frags(P[wave], lane, fr);
#pragma unroll
    for (int t = 0; t < 4; ++t) {                 // x += per_l(agg, w_out)
        v8f o[4];
        gemm_t(fr, wo0, wo1, t, lane, o);
#pragma unroll
        for (int r = 0; r < 8; ++r) {
            int n = r + hi * 8, c = t * 16 + col;
#pragma unroll
            for (int l = 0; l < 4; ++l) xb[wave][n][l][c] += o[l][r];
        }
    }
    __syncthreads();

    if (lane < 16) { nrm[wave][0][lane] = 0.f; nrm[wave][1][lane] = 0.f; }
    __syncthreads();
    for (int i = lane; i < 16 * 64; i += 32) {
        int n = i >> 6, c = i & 63;
        float a0 = xb[wave][n][0][c];
        float a1 = xb[wave][n][1][c], a2 = xb[wave][n][2][c], a3 = xb[wave][n][3][c];
        atomicAdd(&nrm[wave][0][n], a0 * a0);
        atomicAdd(&nrm[wave][1][n], a1 * a1 + a2 * a2 + a3 * a3);
    }
    __syncthreads();
    for (int i = lane; i < 16 * 64; i += 32) {    // panels = eq_norm(x, ln2)
        int n = i >> 6, c = i & 63;
        float i0 = rsqrtf(nrm[wave][0][n] * (1.f / 64.f) + 1e-6f);
        float i1 = rsqrtf(nrm[wave][1][n] * (1.f / 192.f) + 1e-6f);
        P[wave][0][n][c] = f2bf(xb[wave][n][0][c] * i0 * ln2[c]);
        P[wave][1][n][c] = f2bf(xb[wave][n][1][c] * i1 * ln2[64 + c]);
        P[wave][2][n][c] = f2bf(xb[wave][n][2][c] * i1 * ln2[64 + c]);
        P[wave][3][n][c] = f2bf(xb[wave][n][3][c] * i1 * ln2[64 + c]);
    }
    __syncthreads();

    load_frags(P[wave], lane, fr);                // GEMM ffn_w1, gated nonlinearity
#pragma unroll
    for (int t = 0; t < 4; ++t) {
        v8f o[4];
        gemm_t(fr, f10, f11, t, lane, o);
#pragma unroll
        for (int r = 0; r < 8; ++r) {
            int n = r + hi * 8, c = t * 16 + col;
            float gate = sigm(o[0][r]);           // silu(h0)=h0*gate; hl*=gate
            P[wave][0][n][c] = f2bf(o[0][r] * gate);
            P[wave][1][n][c] = f2bf(o[1][r] * gate);
            P[wave][2][n][c] = f2bf(o[2][r] * gate);
            P[wave][3][n][c] = f2bf(o[3][r] * gate);
        }
    }
    __syncthreads();

    load_frags(P[wave], lane, fr);                // GEMM ffn_w2, residual
#pragma unroll
    for (int t = 0; t < 4; ++t) {
        v8f o[4];
        gemm_t(fr, f20, f21, t, lane, o);
#pragma unroll
        for (int r = 0; r < 8; ++r) {
            int n = r + hi * 8, c = t * 16 + col;
#pragma unroll
            for (int l = 0; l < 4; ++l) xb[wave][n][l][c] += o[l][r];
        }
    }
    __syncthreads();
    for (int i = lane; i < 16 * 64; i += 32) {
        int n = i >> 6, c = i & 63;
        float* o = x_out + (size_t)(n0 + n) * 256 + c;
        o[0] = xb[wave][n][0][c];   o[64]  = xb[wave][n][1][c];
        o[128] = xb[wave][n][2][c]; o[192] = xb[wave][n][3][c];
    }
}

// ---------------------------------------------------------------------------
// Edge update: e' = silu([x_src0,x_dst0,e] @ w1 + b1) @ w2 + b2 (K=160 -> 32),
// with w1|w2 fragments TDM-staged into LDS (12KB contiguous).
// ---------------------------------------------------------------------------
__global__ void __launch_bounds__(128)
k_edge_update(const float* __restrict__ x, const float* __restrict__ e_in,
              const int* __restrict__ ei, int E,
              const unsigned* __restrict__ w1p, const float* __restrict__ b1,
              const float* __restrict__ b2, float* __restrict__ e_out) {
    int wave = threadIdx.x >> 5, lane = threadIdx.x & 31;
    int e0 = (blockIdx.x * 4 + wave) * 16;
    __shared__ unsigned short A[4][16][160];
    __shared__ unsigned short Hb[4][16][32];
    __shared__ unsigned wbuf[3072];               // 12KB: w1p(2560dw) | w2p(512dw)
#if HAVE_TDM
    if (threadIdx.x < 32)
        tdm_copy_to_lds(w1p, (unsigned)(size_t)(void*)wbuf, 3072);
#else
    for (int i = threadIdx.x; i < 3072; i += 128) wbuf[i] = w1p[i];
#endif
    int col = lane & 15, hi = (lane >> 4) & 1;
    for (int i = lane; i < 16 * 64; i += 32) {
        int e = i >> 6, c = i & 63;
        int s = ei[e0 + e], d = ei[E + e0 + e];
        A[wave][e][c]      = f2bf(x[(size_t)s * 256 + c]);
        A[wave][e][64 + c] = f2bf(x[(size_t)d * 256 + c]);
    }
    for (int i = lane; i < 16 * 32; i += 32) {
        int e = i >> 5, k = i & 31;
        A[wave][e][128 + k] = f2bf(e_in[(size_t)(e0 + e) * 32 + k]);
    }
    __syncthreads();
    v16bf af[5];
#pragma unroll
    for (int ks = 0; ks < 5; ++ks)
        af[ks] = load_a_frag(&A[wave][0][0], 160, ks * 32, lane);
#pragma unroll
    for (int t = 0; t < 2; ++t) {
        v8f acc = zero8();
#pragma unroll
        for (int ks = 0; ks < 5; ++ks)
            acc = wmma_bf(af[ks], load_b_frag_lds(wbuf, 2, ks, t, lane), acc);
#pragma unroll
        for (int r = 0; r < 8; ++r) {
            int e = r + hi * 8, c = t * 16 + col;
            float v = acc[r] + b1[c];
            Hb[wave][e][c] = f2bf(v * sigm(v));
        }
    }
    __syncthreads();
    v16bf ah = load_a_frag(&Hb[wave][0][0], 32, 0, lane);
#pragma unroll
    for (int t = 0; t < 2; ++t) {
        v8f acc = wmma_bf(ah, load_b_frag_lds(wbuf + 2560, 2, 0, t, lane), zero8());
#pragma unroll
        for (int r = 0; r < 8; ++r) {
            int e = r + hi * 8, c = t * 16 + col;
            e_out[(size_t)(e0 + e) * 32 + c] = acc[r] + b2[c];
        }
    }
}

__global__ void k_gather_res(const float* __restrict__ x, const int* __restrict__ ca,
                             float* __restrict__ res) {
    int i = blockIdx.x * blockDim.x + threadIdx.x;
    if (i >= NRES * 256) return;
    int r = i >> 8, j = i & 255;
    res[i] = x[(size_t)ca[r] * 256 + j];
}

// ---------------------------------------------------------------------------
// Host orchestration.
// ---------------------------------------------------------------------------
extern "C" void kernel_launch(void* const* d_in, const int* in_sizes, int n_in,
                              void* d_out, int out_size, void* d_ws, size_t ws_size,
                              hipStream_t stream) {
    (void)in_sizes; (void)n_in; (void)out_size; (void)ws_size;
    const float* atom_emb = (const float*)d_in[0];
    const float* atom_x   = (const float*)d_in[1];
    const float* X_ca     = (const float*)d_in[2];
    const int* r_ei   = (const int*)d_in[3];
    const int* a_ei   = (const int*)d_in[4];
    const int* res_ei = (const int*)d_in[5];
    const int* ca_idx = (const int*)d_in[6];
    // params are jax-pytree leaves in alphabetical order after the 7 arrays:
    // atom_blocks[4]{alpha_w,edge_b,edge_w,ffn_w1,ffn_w2,ln1_g,ln2_g,w_msg,w_out},
    // edge_updates[4]{b1,b2,w1,w2}, embed_b, embed_w,
    // out_logvar, out_mu, project, res_agg (each 9 leaves).
    auto leaf = [&](int i) -> const float* { return (const float*)d_in[7 + i]; };

    size_t off = 0;
    auto alloc = [&](size_t b) -> void* {
        void* p = (char*)d_ws + off;
        off = (off + b + 255) & ~(size_t)255;
        return p;
    };
    float* x_at  = (float*)alloc((size_t)NATOMS * 256 * 4);
    float* xn    = (float*)alloc((size_t)NATOMS * 256 * 4);
    float* agg   = (float*)alloc((size_t)NATOMS * 256 * 4);
    float* eA    = (float*)alloc((size_t)ERAD * 32 * 4);
    float* eB    = (float*)alloc((size_t)ERAD * 32 * 4);
    float* D1b   = (float*)alloc((size_t)ERAD * 9 * 4);
    float* sbuf  = (float*)alloc((size_t)ERAD * 64 * 4);
    float* elb   = (float*)alloc((size_t)ERAD * 8 * 4);
    unsigned* lmax = (unsigned*)alloc((size_t)NATOMS * 8 * 4);
    float* den   = (float*)alloc((size_t)NATOMS * 8 * 4);
    float* res_x = (float*)alloc((size_t)NRES * 256 * 4);

    auto pack = [&](const float* W, int K, int Nr, int Np) -> unsigned* {
        int ks = (K + 31) / 32, nt = Np / 16;
        int total = ks * nt * 256;
        unsigned* dst = (unsigned*)alloc((size_t)total * 4);
        k_pack_w<<<(total + 255) / 256, 256, 0, stream>>>(W, K, Nr, Np, dst);
        return dst;
    };

    struct BP {
        const float *edge_b, *ln1, *ln2;
        unsigned *ewp, *awp, *wm0, *wm1, *wo0, *wo1, *f10, *f11, *f20, *f21;
    };
    BP bp[8];
    const int base[8] = {0, 9, 18, 27, 81, 72, 63, 54};  // atoms0-3, res_agg, project, mu, logvar
    for (int b = 0; b < 8; ++b) {
        int o = base[b];
        const float* aw = leaf(o + 0);
        bp[b].edge_b = leaf(o + 1);
        const float* ew = leaf(o + 2);
        const float* f1 = leaf(o + 3);
        const float* f2 = leaf(o + 4);
        bp[b].ln1 = leaf(o + 5);
        bp[b].ln2 = leaf(o + 6);
        const float* wm = leaf(o + 7);
        const float* wo = leaf(o + 8);
        bp[b].ewp = pack(ew, 32, 64, 64);
        bp[b].awp = pack(aw, 192, 8, 16);               // N padded 8 -> 16
        // NOTE: wm0/wm1 must stay contiguous (16KB) for the TDM copy.
        bp[b].wm0 = pack(wm, 64, 64, 64);
        bp[b].wm1 = pack(wm + 4096, 64, 64, 64);
        bp[b].wo0 = pack(wo, 64, 64, 64);
        bp[b].wo1 = pack(wo + 4096, 64, 64, 64);
        bp[b].f10 = pack(f1, 64, 64, 64);
        bp[b].f11 = pack(f1 + 4096, 64, 64, 64);
        bp[b].f20 = pack(f2, 64, 64, 64);
        bp[b].f21 = pack(f2 + 4096, 64, 64, 64);
    }
    struct EP { const float *b1, *b2; unsigned *w1p; };
    EP ep[4];
    for (int i = 0; i < 4; ++i) {
        int o = 36 + i * 4;
        ep[i].b1 = leaf(o + 0);
        ep[i].b2 = leaf(o + 1);
        // NOTE: w1p|w2p must stay contiguous (12KB) for the TDM copy.
        ep[i].w1p = pack(leaf(o + 2), 160, 32, 32);
        (void)pack(leaf(o + 3), 32, 32, 32);   // w2p = w1p + 2560 dwords
    }
    const float* embed_b = leaf(52);
    const float* embed_w = leaf(53);

    k_embed<<<NATOMS, 64, 0, stream>>>(atom_emb, embed_w, embed_b, x_at);

    auto run_block = [&](const float* xin, float* xout, int Nn, const int* ei, int E,
                         const float* ef, const BP& p) {
        k_eqnorm<<<Nn, 64, 0, stream>>>(xin, p.ln1, xn);
        k_fill<<<(Nn * 8 + 255) / 256, 256, 0, stream>>>(lmax, 0x007FFFFFu, Nn * 8);
        hipMemsetAsync(den, 0, (size_t)Nn * 8 * 4, stream);
        hipMemsetAsync(agg, 0, (size_t)Nn * 256 * 4, stream);
        k_edge_s_logits<<<E / 64, 128, 0, stream>>>(ef, xn, ei, E, p.ewp, p.edge_b,
                                                    p.awp, sbuf, elb, lmax);
        k_edge_expsum<<<(E * 8 + 255) / 256, 256, 0, stream>>>(elb, lmax, ei, E, den);
        k_edge_message<<<E / 64, 128, 0, stream>>>(xn, ei, E, D1b, sbuf, elb, den,
                                                   p.wm0, p.wm1, agg);
        k_node_update<<<Nn / 64, 128, 0, stream>>>(xin, xout, agg, p.wo0, p.wo1,
                                                   p.ln2, p.f10, p.f11, p.f20, p.f21);
    };

    // --- atomic radius interaction layers ---
    k_edge_geom<<<(ERAD + 127) / 128, 128, 0, stream>>>(atom_x, r_ei, ERAD, 0, D1b, eA);
    float* ecur = eA;
    float* enext = eB;
    for (int l = 0; l < NLAY; ++l) {
        run_block(x_at, x_at, NATOMS, r_ei, ERAD, ecur, bp[l]);
        k_edge_update<<<ERAD / 64, 128, 0, stream>>>(x_at, ecur, r_ei, ERAD,
                                                     ep[l].w1p, ep[l].b1,
                                                     ep[l].b2, enext);
        float* t = ecur; ecur = enext; enext = t;
    }
    // --- aggregate atoms to CA ---
    k_edge_geom<<<(NATOMS + 127) / 128, 128, 0, stream>>>(atom_x, a_ei, NATOMS, 0, D1b, enext);
    run_block(x_at, x_at, NATOMS, a_ei, NATOMS, enext, bp[4]);
    k_gather_res<<<(NRES * 256 + 255) / 256, 256, 0, stream>>>(x_at, ca_idx, res_x);
    // --- residue knn graph: project, mu, logvar ---
    k_edge_geom<<<(ERES + 127) / 128, 128, 0, stream>>>(X_ca, res_ei, ERES, 1, D1b, enext);
    run_block(res_x, res_x, NRES, res_ei, ERES, enext, bp[5]);
    float* outp = (float*)d_out;
    run_block(res_x, outp, NRES, res_ei, ERES, enext, bp[6]);
    run_block(res_x, outp + (size_t)NRES * 256, NRES, res_ei, ERES, enext, bp[7]);
}
